// PCLayer_72172630442448
// MI455X (gfx1250) — matrix operand order repaired and checked by reading, not compiled, not run on hardware
//
#include <hip/hip_runtime.h>
#include <hip/hip_bf16.h>

// ---------------------------------------------------------------------------
// Predictive-coding T-step loop on MI455X (gfx1250).
// Compute-bound (~580 GFLOP, working set ~85MB fits in 192MB L2), so the
// GEMMs run on v_wmma_f32_16x16x32_bf16 with fp32 accumulation; fp32 master
// copies of x / W / b are kept for the small-lr updates.
// ---------------------------------------------------------------------------

#define LOCAL_LR 0.001f
#define CLAMP_V  1.0f

typedef __bf16 v16bf __attribute__((ext_vector_type(16)));
typedef float  v8f   __attribute__((ext_vector_type(8)));
typedef unsigned short u16;

__device__ __forceinline__ u16 f2bf(float f) {
  unsigned int u = __float_as_uint(f);
  u += 0x7FFFu + ((u >> 16) & 1u);          // round-to-nearest-even
  return (u16)(u >> 16);
}
__device__ __forceinline__ float bf2f(u16 h) {
  return __uint_as_float(((unsigned int)h) << 16);
}

union FragAB {
  uint4 u[2];
  v16bf v;
};

enum { MODE_ERR = 0, MODE_XUP = 1, MODE_DW = 2 };

// ---------------------------------------------------------------------------
// Tiled WMMA GEMM: C[M,N] = A[M,K] * B[K,N]  (A,B bf16; acc fp32)
//   block tile 128x128, BK=32, 256 threads = 8 waves (4 in M x 2 in N),
//   each wave owns 32x64 = 2x4 WMMA tiles.  LDS double-buffered, pitch 40
//   (80B rows: 16B-aligned b128 fragment reads, bank-conflict-free).
//   TRANSA stages A[m][k] = src[k*lda + m]  (for err^T in the dW GEMM).
//   bf16 pair-packing done with v_perm_b32 (1 VALU per packed dword).
// ---------------------------------------------------------------------------
template <int MODE, bool TRANSA>
__global__ __launch_bounds__(256) void pc_wmma_gemm(
    const u16* __restrict__ A, int lda,
    const u16* __restrict__ B, int ldb,
    int K, int ldc,
    const float* __restrict__ e_tgt, const float* __restrict__ e_bias,
    u16* __restrict__ e_err,
    const float* __restrict__ x_old, float* __restrict__ x_new,
    u16* __restrict__ xb_new,
    float* __restrict__ w_f, u16* __restrict__ w_b,
    u16* __restrict__ w_tb, int ldwt)
{
  __shared__ u16 As[2][128][40];
  __shared__ u16 Bs[2][128][40];

  const int t  = threadIdx.x;
  const int m0 = blockIdx.y * 128;
  const int n0 = blockIdx.x * 128;

  const int lane  = t & 31;
  const int wv    = t >> 5;
  const int waveM = wv >> 1;   // 0..3  (32-row strip)
  const int waveN = wv & 1;    // 0..1  (64-col strip)
  const int lm    = lane & 15;
  const int hi    = lane >> 4;

  v8f acc[2][4];
  const v8f vzero = {0.f, 0.f, 0.f, 0.f, 0.f, 0.f, 0.f, 0.f};
#pragma unroll
  for (int i = 0; i < 2; ++i)
#pragma unroll
    for (int j = 0; j < 4; ++j) acc[i][j] = vzero;

  const int nk = K >> 5;

  // ---- per-thread staging source pointers, advanced by a constant stride
  // A (no-trans): thread covers row m = t/2, k half = (t&1)*16
  // A (trans)   : thread covers k rows 2kp,2kp+1 (kp=t/16), 8 cols o0=(t%16)*8
  // B           : thread covers k rows 2kp,2kp+1, 8 cols nn=(t%16)*8
  const int kp   = t >> 4;
  const int run8 = (t & 15) << 3;
  const u16* pA;
  ptrdiff_t stepA;
  if (!TRANSA) {
    pA    = A + (size_t)(m0 + (t >> 1)) * lda + ((t & 1) << 4);
    stepA = 32;
  } else {
    pA    = A + (size_t)(2 * kp) * lda + (m0 + run8);
    stepA = (ptrdiff_t)32 * lda;
  }
  const u16* pB = B + (size_t)(2 * kp) * ldb + (n0 + run8);
  const ptrdiff_t stepB = (ptrdiff_t)32 * ldb;

  auto stage = [&](int buf) {
    // ---- A tile -> As[m][k]
    if (!TRANSA) {
      const int m = t >> 1;
      const int k = (t & 1) << 4;
      uint4 q0 = *(const uint4*)(pA);
      uint4 q1 = *(const uint4*)(pA + 8);
      *(uint4*)(&As[buf][m][k])     = q0;
      *(uint4*)(&As[buf][m][k + 8]) = q1;
    } else {
      // As[o][kk] = A[(k0+kk)*lda + (m0+o)]  : transpose on LDS store
      uint4 q0 = *(const uint4*)(pA);
      uint4 q1 = *(const uint4*)(pA + lda);
      const unsigned int* w0 = (const unsigned int*)&q0;
      const unsigned int* w1 = (const unsigned int*)&q1;
#pragma unroll
      for (int w = 0; w < 4; ++w) {
        unsigned int plo = __builtin_amdgcn_perm(w1[w], w0[w], 0x05040100u);
        unsigned int phi = __builtin_amdgcn_perm(w1[w], w0[w], 0x07060302u);
        *(unsigned int*)(&As[buf][run8 + 2 * w][2 * kp])     = plo;
        *(unsigned int*)(&As[buf][run8 + 2 * w + 1][2 * kp]) = phi;
      }
    }
    pA += stepA;
    // ---- B tile -> Bs[n][k]  (source row-major [K][N], transpose on store)
    {
      uint4 q0 = *(const uint4*)(pB);
      uint4 q1 = *(const uint4*)(pB + ldb);
      const unsigned int* w0 = (const unsigned int*)&q0;
      const unsigned int* w1 = (const unsigned int*)&q1;
#pragma unroll
      for (int w = 0; w < 4; ++w) {
        unsigned int plo = __builtin_amdgcn_perm(w1[w], w0[w], 0x05040100u);
        unsigned int phi = __builtin_amdgcn_perm(w1[w], w0[w], 0x07060302u);
        *(unsigned int*)(&Bs[buf][run8 + 2 * w][2 * kp])     = plo;
        *(unsigned int*)(&Bs[buf][run8 + 2 * w + 1][2 * kp]) = phi;
      }
      pB += stepB;
    }
  };

  stage(0);
  __syncthreads();

  for (int c = 0; c < nk; ++c) {
    const int cur = c & 1;
    if (c + 1 < nk) stage(cur ^ 1);

    FragAB fa[2], fb[4];
#pragma unroll
    for (int sm = 0; sm < 2; ++sm) {
      const int row = waveM * 32 + sm * 16 + lm;
      // A frag (ISA 7.12.2): lane hi half: K = hi*8 + j (j<8), hi*8+16+j-8 (j>=8)
      fa[sm].u[0] = *(const uint4*)(&As[cur][row][hi * 8]);
      fa[sm].u[1] = *(const uint4*)(&As[cur][row][hi * 8 + 16]);
    }
#pragma unroll
    for (int sn = 0; sn < 4; ++sn) {
      const int col = waveN * 64 + sn * 16 + lm;
      // B frag: same K order as A: K = hi*8 + j (j<8), hi*8+16+j-8 (j>=8), N = lane%16
      fb[sn].u[0] = *(const uint4*)(&Bs[cur][col][hi * 8]);
      fb[sn].u[1] = *(const uint4*)(&Bs[cur][col][hi * 8 + 16]);
    }
#pragma unroll
    for (int sm = 0; sm < 2; ++sm)
#pragma unroll
      for (int sn = 0; sn < 4; ++sn)
        acc[sm][sn] = __builtin_amdgcn_wmma_f32_16x16x32_bf16(
            false, fa[sm].v, false, fb[sn].v, (short)0, acc[sm][sn],
            false, false);

    __syncthreads();
  }

  // ---- epilogue: C/D layout M = r + 8*hi, N = lane%16
#pragma unroll
  for (int sm = 0; sm < 2; ++sm) {
#pragma unroll
    for (int sn = 0; sn < 4; ++sn) {
      const int nn = n0 + waveN * 64 + sn * 16 + lm;
#pragma unroll
      for (int r = 0; r < 8; ++r) {
        const int mm = m0 + waveM * 32 + sm * 16 + hi * 8 + r;
        const float v = acc[sm][sn][r];
        const size_t idx = (size_t)mm * ldc + nn;
        if (MODE == MODE_ERR) {
          float e = e_tgt[idx] - (v + e_bias[nn]);
          e_err[idx] = f2bf(e);
        } else if (MODE == MODE_XUP) {
          float xv = x_old[idx] + LOCAL_LR * v;
          xv = fminf(CLAMP_V, fmaxf(-CLAMP_V, xv));
          x_new[idx]  = xv;
          xb_new[idx] = f2bf(xv);
        } else {  // MODE_DW
          float w = w_f[idx] + LOCAL_LR * v;
          w_f[idx] = w;
          u16 h = f2bf(w);
          w_b[idx] = h;
          w_tb[(size_t)nn * ldwt + mm] = h;
        }
      }
    }
  }
}

// ---------------------------------------------------------------------------
// small helper kernels
// ---------------------------------------------------------------------------
__global__ void pc_init_x(const float* __restrict__ x0, float* __restrict__ xf,
                          u16* __restrict__ xb, int n) {
  int i = blockIdx.x * 256 + threadIdx.x;
  if (i < n) { float v = x0[i]; xf[i] = v; xb[i] = f2bf(v); }
}

__global__ void pc_init_w(const float* __restrict__ W, float* __restrict__ wf,
                          u16* __restrict__ wb, u16* __restrict__ wtb, int n) {
  int i = blockIdx.x * 256 + threadIdx.x;
  if (i < n) {
    float w = W[i];
    wf[i] = w;
    u16 h = f2bf(w);
    wb[i] = h;
    int o = i / 768, ii = i - o * 768;
    wtb[(size_t)ii * 3072 + o] = h;   // W^T copy for mu = x @ W^T
  }
}

__global__ void pc_init_b(const float* __restrict__ b, float* __restrict__ bf_) {
  int i = blockIdx.x * 256 + threadIdx.x;
  if (i < 3072) bf_[i] = b[i];
}

// deterministic bias column-sum: 16 row-partials, fixed-order reduce
__global__ void pc_bias_partial(const u16* __restrict__ err, float* __restrict__ part) {
  int n  = blockIdx.x * 256 + threadIdx.x;   // 12 blocks x 256 cols
  int mb = blockIdx.y * 256;                 // 16 row strips
  float s = 0.f;
  for (int m = 0; m < 256; ++m) s += bf2f(err[(size_t)(mb + m) * 3072 + n]);
  part[(size_t)blockIdx.y * 3072 + n] = s;
}

__global__ void pc_bias_apply(float* __restrict__ bf_, const float* __restrict__ part) {
  int n = blockIdx.x * 256 + threadIdx.x;
  float s = 0.f;
  for (int i = 0; i < 16; ++i) s += part[(size_t)i * 3072 + n];
  bf_[n] += LOCAL_LR * s;
}

// ---------------------------------------------------------------------------
extern "C" void kernel_launch(void* const* d_in, const int* in_sizes, int n_in,
                              void* d_out, int out_size, void* d_ws, size_t ws_size,
                              hipStream_t stream) {
  const float* target = (const float*)d_in[0];  // [4096,3072]
  const float* x0     = (const float*)d_in[1];  // [4096,768]
  const float* W      = (const float*)d_in[2];  // [3072,768]
  const float* b      = (const float*)d_in[3];  // [3072]

  const int M = 4096, HI = 768, HO = 3072;
  const size_t NX = (size_t)M * HI, NE = (size_t)M * HO, NW = (size_t)HO * HI;

  char* p = (char*)d_ws;
  auto carve = [&](size_t bytes) {
    char* r = p;
    p += (bytes + 255) & ~(size_t)255;
    return (void*)r;
  };
  float* xf[2]; u16* xb[2];
  xf[0] = (float*)carve(NX * 4);
  xf[1] = (float*)carve(NX * 4);
  xb[0] = (u16*)carve(NX * 2);
  xb[1] = (u16*)carve(NX * 2);
  u16*   errb = (u16*)carve(NE * 2);
  float* Wf   = (float*)carve(NW * 4);
  u16*   Wb   = (u16*)carve(NW * 2);
  u16*   Wtb  = (u16*)carve(NW * 2);
  float* bfv  = (float*)carve((size_t)HO * 4);
  float* part = (float*)carve((size_t)16 * HO * 4);

  if ((size_t)(p - (char*)d_ws) > ws_size) return;  // ws too small: bail

  pc_init_x<<<(int)(NX / 256), 256, 0, stream>>>(x0, xf[0], xb[0], (int)NX);
  pc_init_w<<<(int)(NW / 256), 256, 0, stream>>>(W, Wf, Wb, Wtb, (int)NW);
  pc_init_b<<<12, 256, 0, stream>>>(b, bfv);

  int cur = 0;
  for (int ts = 0; ts < 10; ++ts) {
    const int nxt = cur ^ 1;

    // err = target - (x @ W^T + b)   [M=4096, N=3072, K=768]
    pc_wmma_gemm<MODE_ERR, false><<<dim3(HO / 128, M / 128), 256, 0, stream>>>(
        xb[cur], HI, Wtb, HO, HI, HO,
        target, bfv, errb,
        nullptr, nullptr, nullptr,
        nullptr, nullptr, nullptr, 0);

    pc_bias_partial<<<dim3(HO / 256, M / 256), 256, 0, stream>>>(errb, part);

    // x_new = clamp(x + lr * err @ W)   [M=4096, N=768, K=3072]
    float* xdst = (ts == 9) ? (float*)d_out : xf[nxt];
    pc_wmma_gemm<MODE_XUP, false><<<dim3(HI / 128, M / 128), 256, 0, stream>>>(
        errb, HO, Wb, HI, HO, HI,
        nullptr, nullptr, nullptr,
        xf[cur], xdst, xb[nxt],
        nullptr, nullptr, nullptr, 0);

    // W += lr * err^T @ x_old   [M=3072, N=768, K=4096]  (TRANSA staging)
    pc_wmma_gemm<MODE_DW, true><<<dim3(HI / 128, HO / 128), 256, 0, stream>>>(
        errb, HO, xb[cur], HI, M, HI,
        nullptr, nullptr, nullptr,
        nullptr, nullptr, nullptr,
        Wf, Wb, Wtb, HO);

    pc_bias_apply<<<12, 256, 0, stream>>>(bfv, part);

    cur = nxt;
  }

  (void)in_sizes; (void)n_in; (void)out_size;
}
